// SignalLongformerEncoder_58677843198502
// MI455X (gfx1250) — matrix operand order repaired and checked by reading, not compiled, not run on hardware
//
#include <hip/hip_runtime.h>
#include <math.h>

// ---------------------------------------------------------------------------
// SignalLongformerEncoder for MI455X (gfx1250), wave32 + WMMA bf16.
//
// Pipeline:
//   conv1 (scalar fp32)  -> GN+gelu
//   conv2 as im2col GEMM (bf16 WMMA) -> GN+gelu
//   conv3 as im2col GEMM (bf16 WMMA) -> GN+gelu -> +PE  => x[1152,800]
//   8x { QKV gemm, windowed attention, out-proj gemm, add+LN,
//        FFN gemm(gelu), FFN gemm, add+LN }   (all GEMMs = bf16 WMMA)
//   final gemm -> d_out
//
// GEMM: each wave computes a 32x64 tile (8x v_wmma_f32_16x16x32_bf16 per
// K-step), B tiles double-buffered through LDS (1 barrier per K-step).
// ---------------------------------------------------------------------------

typedef __attribute__((ext_vector_type(16))) __bf16 v16bf;
typedef __attribute__((ext_vector_type(8)))  float  v8f;

#define BATCH 64
#define SEQ   18
#define DIM   800
#define NHEAD 8
#define HDIM  100
#define FFD   3200
#define NLAY  8

__device__ __forceinline__ __bf16 f2bf(float f) {
  union { float f; unsigned u; } in; in.f = f;
  unsigned u = in.u;
  unsigned r = u + 0x7FFFu + ((u >> 16) & 1u);   // round-to-nearest-even
  unsigned short h = (unsigned short)(r >> 16);
  union { unsigned short s; __bf16 b; } out; out.s = h;
  return out.b;
}

__device__ __forceinline__ float gelu_f(float v) {
  float v3 = v * v * v;
  return 0.5f * v * (1.0f + tanhf(0.7978845608028654f * (v + 0.044715f * v3)));
}

// A-fragment K position for (lane, element e) within a 16x32 bf16 tile
// (CDNA5 ISA 7.12.2: 16-bit A-matrix layout).
__device__ __forceinline__ int a_klocal(int lane, int e) {
  int v = e >> 1, p = e & 1;
  return ((v & 4) ? 16 : 0) + ((lane >> 4) << 3) + ((v & 3) << 1) + p;
}

// ---------------------------------------------------------------------------
// conv1: src[64,12,9000] -> out[64,8,1800], stride 5, pad 100, k 201 (+bias)
// ---------------------------------------------------------------------------
__global__ __launch_bounds__(256) void conv1_kernel(
    const float* __restrict__ src, const float* __restrict__ w,
    const float* __restrict__ bias, float* __restrict__ out) {
  int idx = blockIdx.x * 256 + threadIdx.x;
  if (idx >= BATCH * 8 * 1800) return;
  int l = idx % 1800;
  int c = (idx / 1800) % 8;
  int b = idx / (1800 * 8);
  const float* sp = src + (size_t)b * 12 * 9000;
  const float* wp = w + (size_t)c * 12 * 201;
  float acc = bias[c];
  int l0 = l * 5 - 100;
  for (int ci = 0; ci < 12; ++ci) {
    const float* s = sp + ci * 9000;
    const float* wk = wp + ci * 201;
    for (int kk = 0; kk < 201; ++kk) {
      int li = l0 + kk;
      if (li >= 0 && li < 9000) acc = fmaf(s[li], wk[kk], acc);
    }
  }
  out[idx] = acc;   // layout [b, c, l]
}

// ---------------------------------------------------------------------------
// GroupNorm(1,C) + GELU in place; one block per sample (n elems each).
// channel index = (i / cdiv) % C
// ---------------------------------------------------------------------------
__global__ __launch_bounds__(256) void gn_gelu_kernel(
    float* __restrict__ x, const float* __restrict__ g,
    const float* __restrict__ bta, int C, int cdiv, int n) {
  float* p = x + (size_t)blockIdx.x * n;
  __shared__ float s_sum[256], s_sq[256];
  float ls = 0.f, lq = 0.f;
  for (int i = threadIdx.x; i < n; i += 256) { float v = p[i]; ls += v; lq += v * v; }
  s_sum[threadIdx.x] = ls; s_sq[threadIdx.x] = lq;
  __syncthreads();
  for (int off = 128; off > 0; off >>= 1) {
    if ((int)threadIdx.x < off) {
      s_sum[threadIdx.x] += s_sum[threadIdx.x + off];
      s_sq[threadIdx.x]  += s_sq[threadIdx.x + off];
    }
    __syncthreads();
  }
  float mean = s_sum[0] / (float)n;
  float var  = s_sq[0] / (float)n - mean * mean;
  float inv  = rsqrtf(var + 1e-5f);
  for (int i = threadIdx.x; i < n; i += 256) {
    int c = (i / cdiv) % C;
    float v = (p[i] - mean) * inv * g[c] + bta[c];
    p[i] = gelu_f(v);
  }
}

// ---------------------------------------------------------------------------
// Generic packers into WMMA fragment order (tile-major [tm|tn][tk][lane][e]).
// ---------------------------------------------------------------------------
__global__ __launch_bounds__(256) void pack_a_kernel(
    const float* __restrict__ A, __bf16* __restrict__ Ap,
    int M, int K, int KT, int total) {
  int idx = blockIdx.x * 256 + threadIdx.x;
  if (idx >= total) return;
  int e = idx & 15, lane = (idx >> 4) & 31, tile = idx >> 9;
  int tk = tile % KT, tm = tile / KT;
  int m = tm * 16 + (lane & 15);
  int k = tk * 32 + a_klocal(lane, e);
  float val = (m < M && k < K) ? A[(size_t)m * K + k] : 0.f;
  Ap[idx] = f2bf(val);
}

__global__ __launch_bounds__(256) void pack_b_kernel(
    const float* __restrict__ W, __bf16* __restrict__ Bp,
    int N, int K, int KT, int total) {
  int idx = blockIdx.x * 256 + threadIdx.x;
  if (idx >= total) return;
  int e = idx & 15, lane = (idx >> 4) & 31, tile = idx >> 9;
  int tk = tile % KT, tn = tile / KT;
  int k = tk * 32 + a_klocal(lane, e);   // B 32x16: lanes carry N, elements carry K (same order as A)
  int n = tn * 16 + (lane & 15);
  float val = (n < N && k < K) ? W[(size_t)n * K + k] : 0.f;
  Bp[idx] = f2bf(val);
}

// im2col packers for conv2 / conv3 (produce A fragments directly)
__global__ __launch_bounds__(256) void im2col2_kernel(
    const float* __restrict__ c1, __bf16* __restrict__ Ap, int total) {
  int idx = blockIdx.x * 256 + threadIdx.x;
  if (idx >= total) return;
  int e = idx & 15, lane = (idx >> 4) & 31, tile = idx >> 9;
  const int KT = 26;
  int tk = tile % KT, tm = tile / KT;
  int m = tm * 16 + (lane & 15);          // m = b*180 + t
  int k = tk * 32 + a_klocal(lane, e);
  float val = 0.f;
  if (k < 808) {
    int c = k / 101, kk = k % 101;
    int b = m / 180, t = m % 180;
    int li = t * 10 - 50 + kk;
    if (li >= 0 && li < 1800) val = c1[((size_t)b * 8 + c) * 1800 + li];
  }
  Ap[idx] = f2bf(val);
}

__global__ __launch_bounds__(256) void im2col3_kernel(
    const float* __restrict__ c2, __bf16* __restrict__ Ap, int total) {
  int idx = blockIdx.x * 256 + threadIdx.x;
  if (idx >= total) return;
  int e = idx & 15, lane = (idx >> 4) & 31, tile = idx >> 9;
  const int KT = 128;
  int tk = tile % KT, tm = tile / KT;
  int m = tm * 16 + (lane & 15);          // m = b*18 + t
  int k = tk * 32 + a_klocal(lane, e);
  float val = 0.f;
  if (k < 4080) {
    int c = k / 51, kk = k % 51;
    int b = m / 18, t = m % 18;
    int li = t * 10 - 25 + kk;
    if (li >= 0 && li < 180) val = c2[(size_t)b * 14400 + (size_t)li * 80 + c];
  }
  Ap[idx] = f2bf(val);
}

// ---------------------------------------------------------------------------
// bf16 WMMA GEMM: C[M,N] = A[M,K] * W[N,K]^T + bias (mode 1: + gelu).
// Block = 4 waves; wave w computes a 32x64 tile:
//   m-tiles  (blockIdx.y*8 + 2w, +1), n-tiles (blockIdx.x*4 .. +3).
// B n-tiles double-buffered through LDS: one barrier per K-step,
// next tile is global-loaded into registers while WMMAs consume LDS.
// ---------------------------------------------------------------------------
__global__ __launch_bounds__(128) void gemm_bf16_kernel(
    const __bf16* __restrict__ Ap, const __bf16* __restrict__ Bp,
    const float* __restrict__ bias, float* __restrict__ C,
    int MT, int NT, int KT, int M, int N, int mode) {
  __shared__ __bf16 Bs[2][4 * 512];
  const int lane = threadIdx.x & 31;
  const int wave = threadIdx.x >> 5;
  const int tm0 = blockIdx.y * 8 + wave * 2;     // two m-tiles per wave
  const int tn0 = blockIdx.x * 4;

  // clamped A pointers (avoid OOB reads; epilogue guards with raw indices)
  int tma = tm0     < MT ? tm0     : MT - 1;
  int tmb = tm0 + 1 < MT ? tm0 + 1 : MT - 1;
  const __bf16* aptr0 = Ap + (((size_t)tma * KT) << 9) + (size_t)lane * 16;
  const __bf16* aptr1 = Ap + (((size_t)tmb * KT) << 9) + (size_t)lane * 16;
  const bool bvalid = (tn0 + wave) < NT;
  const __bf16* bptr = Bp + (((size_t)(tn0 + wave) * KT) << 9) + (size_t)lane * 16;

  v8f acc[8] = {};   // [mi*4 + j]

  // prime buffer 0 with K-tile 0
  {
    v16bf bv = {};
    if (bvalid) bv = *(const v16bf*)bptr;
    *(v16bf*)(&Bs[0][wave * 512 + lane * 16]) = bv;
  }
  __syncthreads();

  for (int tk = 0; tk < KT; ++tk) {
    const int cur = tk & 1;

    // global-load next B tile into registers (overlaps with WMMA below)
    v16bf nxt = {};
    if (tk + 1 < KT && bvalid)
      nxt = *(const v16bf*)(bptr + ((size_t)(tk + 1) << 9));

    v16bf a0 = *(const v16bf*)(aptr0 + ((size_t)tk << 9));
    v16bf a1 = *(const v16bf*)(aptr1 + ((size_t)tk << 9));
    if (tk + 1 < KT) {
      __builtin_prefetch(aptr0 + ((size_t)(tk + 1) << 9), 0, 1);  // global_prefetch_b8
      __builtin_prefetch(aptr1 + ((size_t)(tk + 1) << 9), 0, 1);
    }

    const __bf16* bs = &Bs[cur][lane * 16];
#pragma unroll
    for (int j = 0; j < 4; ++j) {
      v16bf b = *(const v16bf*)(bs + j * 512);
      acc[j]     = __builtin_amdgcn_wmma_f32_16x16x32_bf16(false, a0, false, b, (short)0, acc[j],     false, false);
      acc[4 + j] = __builtin_amdgcn_wmma_f32_16x16x32_bf16(false, a1, false, b, (short)0, acc[4 + j], false, false);
    }

    if (tk + 1 < KT)
      *(v16bf*)(&Bs[cur ^ 1][wave * 512 + lane * 16]) = nxt;
    __syncthreads();   // writes to Bs[cur^1] visible; Bs[cur] free for reuse
  }

  // Epilogue. C/D layout: n = lane&15, m = r + 8*(lane>>4).
  const int nlo = lane & 15;
  const int mhi = (lane >> 4) << 3;
#pragma unroll
  for (int mi = 0; mi < 2; ++mi) {
    int tm = tm0 + mi;
    if (tm >= MT) continue;
#pragma unroll
    for (int j = 0; j < 4; ++j) {
      int tn = tn0 + j;
      if (tn >= NT) continue;
      int n = tn * 16 + nlo;
      if (n >= N) continue;
      float bv = bias[n];
#pragma unroll
      for (int r = 0; r < 8; ++r) {
        int m = tm * 16 + mhi + r;
        if (m >= M) continue;
        float v = acc[mi * 4 + j][r] + bv;
        if (mode == 1) v = gelu_f(v);
        C[(size_t)m * N + n] = v;
      }
    }
  }
}

// ---------------------------------------------------------------------------
// Positional encoding add (in place) on x[1152,800]
// ---------------------------------------------------------------------------
__global__ __launch_bounds__(256) void pe_add_kernel(float* __restrict__ x) {
  int idx = blockIdx.x * 256 + threadIdx.x;
  if (idx >= BATCH * SEQ * DIM) return;
  int d = idx % DIM;
  int s = (idx / DIM) % SEQ;
  int i2 = d >> 1;
  float div = __expf((float)(2 * i2) * (-9.2103403719761836f / (float)DIM));
  float a = (float)s * div;
  x[idx] += (d & 1) ? __cosf(a) : __sinf(a);
}

// ---------------------------------------------------------------------------
// Windowed attention: one block per (b, h). S=18, HD=100.
// ---------------------------------------------------------------------------
__global__ __launch_bounds__(128) void attn_kernel(
    const float* __restrict__ qkv, float* __restrict__ att) {
  int bh = blockIdx.x;
  int b = bh >> 3, hh = bh & 7;
  __shared__ float q[SEQ][HDIM], k[SEQ][HDIM], v[SEQ][HDIM];
  __shared__ float sc[SEQ][SEQ], rsum[SEQ];

  for (int i = threadIdx.x; i < SEQ * HDIM; i += 128) {
    int s = i / HDIM, d = i % HDIM;
    size_t base = ((size_t)(b * SEQ + s)) * (3 * DIM) + hh * HDIM + d;
    q[s][d] = qkv[base];
    k[s][d] = qkv[base + DIM];
    v[s][d] = qkv[base + 2 * DIM];
  }
  __syncthreads();

  for (int i = threadIdx.x; i < SEQ * SEQ; i += 128) {
    int si = i / SEQ, sj = i % SEQ;
    int diff = si - sj; if (diff < 0) diff = -diff;
    bool masked = (diff > 2) && (si != 0) && (sj != 0);
    float s = -1e30f;
    if (!masked) {
      s = 0.f;
      for (int d = 0; d < HDIM; ++d) s = fmaf(q[si][d], k[sj][d], s);
      s *= 0.1f;   // 1/sqrt(100)
    }
    sc[si][sj] = s;
  }
  __syncthreads();

  if ((int)threadIdx.x < SEQ) {
    int si = threadIdx.x;
    float mx = -1e30f;
    for (int j = 0; j < SEQ; ++j) mx = fmaxf(mx, sc[si][j]);
    float sum = 0.f;
    for (int j = 0; j < SEQ; ++j) { float e = __expf(sc[si][j] - mx); sc[si][j] = e; sum += e; }
    rsum[si] = 1.f / sum;
  }
  __syncthreads();

  for (int i = threadIdx.x; i < SEQ * HDIM; i += 128) {
    int s = i / HDIM, d = i % HDIM;
    float o = 0.f;
    for (int j = 0; j < SEQ; ++j) o = fmaf(sc[s][j], v[j][d], o);
    att[((size_t)(b * SEQ + s)) * DIM + hh * HDIM + d] = o * rsum[s];
  }
}

// ---------------------------------------------------------------------------
// x = LayerNorm(x + r) * g + b, per row of 800; one block per row.
// ---------------------------------------------------------------------------
__global__ __launch_bounds__(256) void add_ln_kernel(
    float* __restrict__ x, const float* __restrict__ r,
    const float* __restrict__ g, const float* __restrict__ bta) {
  int row = blockIdx.x;
  float* xp = x + (size_t)row * DIM;
  const float* rp = r + (size_t)row * DIM;
  __shared__ float buf[DIM];
  __shared__ float s_sum[256], s_sq[256];
  float ls = 0.f, lq = 0.f;
  for (int i = threadIdx.x; i < DIM; i += 256) {
    float v = xp[i] + rp[i];
    buf[i] = v; ls += v; lq += v * v;
  }
  s_sum[threadIdx.x] = ls; s_sq[threadIdx.x] = lq;
  __syncthreads();
  for (int off = 128; off > 0; off >>= 1) {
    if ((int)threadIdx.x < off) {
      s_sum[threadIdx.x] += s_sum[threadIdx.x + off];
      s_sq[threadIdx.x]  += s_sq[threadIdx.x + off];
    }
    __syncthreads();
  }
  float mean = s_sum[0] * (1.f / DIM);
  float var  = s_sq[0] * (1.f / DIM) - mean * mean;
  float inv  = rsqrtf(var + 1e-5f);
  for (int i = threadIdx.x; i < DIM; i += 256) {
    xp[i] = (buf[i] - mean) * inv * g[i] + bta[i];
  }
}

// ---------------------------------------------------------------------------
// Host launch
// ---------------------------------------------------------------------------
extern "C" void kernel_launch(void* const* d_in, const int* in_sizes, int n_in,
                              void* d_out, int out_size, void* d_ws, size_t ws_size,
                              hipStream_t stream) {
  const float* src    = (const float*)d_in[0];
  const float* cw1    = (const float*)d_in[1];
  const float* cb1    = (const float*)d_in[2];
  const float* gn1_g  = (const float*)d_in[3];
  const float* gn1_b  = (const float*)d_in[4];
  const float* cw2    = (const float*)d_in[5];
  const float* cb2    = (const float*)d_in[6];
  const float* gn2_g  = (const float*)d_in[7];
  const float* gn2_b  = (const float*)d_in[8];
  const float* cw3    = (const float*)d_in[9];
  const float* cb3    = (const float*)d_in[10];
  const float* gn3_g  = (const float*)d_in[11];
  const float* gn3_b  = (const float*)d_in[12];
  const float* ipw    = (const float*)d_in[13];
  const float* ipb    = (const float*)d_in[14];
  const float* aow    = (const float*)d_in[15];
  const float* aob    = (const float*)d_in[16];
  const float* l1w    = (const float*)d_in[17];
  const float* l1b    = (const float*)d_in[18];
  const float* l2w    = (const float*)d_in[19];
  const float* l2b    = (const float*)d_in[20];
  const float* ln1g   = (const float*)d_in[21];
  const float* ln1b   = (const float*)d_in[22];
  const float* ln2g   = (const float*)d_in[23];
  const float* ln2b   = (const float*)d_in[24];
  const float* outw   = (const float*)d_in[25];
  const float* outb   = (const float*)d_in[26];
  float* out = (float*)d_out;

  // --- workspace carve (all 256B aligned) ---
  char* p = (char*)d_ws;
  auto carve = [&](size_t bytes) -> void* {
    void* r = (void*)p;
    p += (bytes + 255) & ~(size_t)255;
    return r;
  };
  float*  c1   = (float*) carve((size_t)BATCH * 8 * 1800 * 4);     // [b,c,l]
  float*  c2   = (float*) carve((size_t)BATCH * 180 * 80 * 4);     // [b,t,o]
  float*  x    = (float*) carve((size_t)1152 * DIM * 4);
  float*  qkv  = (float*) carve((size_t)1152 * 3 * DIM * 4);
  float*  att  = (float*) carve((size_t)1152 * DIM * 4);
  float*  tmp  = (float*) carve((size_t)1152 * DIM * 4);           // aout / f
  float*  hbuf = (float*) carve((size_t)1152 * FFD * 4);
  __bf16* Apk  = (__bf16*)carve((size_t)720 * 26 * 512 * 2);       // max A pack
  __bf16* Bpk  = (__bf16*)carve((size_t)50 * 128 * 512 * 2);       // max B pack

  auto packA = [&](const float* A, int M, int K) {
    int MT = (M + 15) / 16, KT = (K + 31) / 32;
    int total = MT * KT * 512;
    pack_a_kernel<<<(total + 255) / 256, 256, 0, stream>>>(A, Apk, M, K, KT, total);
  };
  auto packB = [&](const float* W, int N, int K) {
    int NT = (N + 15) / 16, KT = (K + 31) / 32;
    int total = NT * KT * 512;
    pack_b_kernel<<<(total + 255) / 256, 256, 0, stream>>>(W, Bpk, N, K, KT, total);
  };
  auto gemm = [&](const float* bias, float* C, int M, int N, int K, int mode) {
    int MT = (M + 15) / 16, NT = (N + 15) / 16, KT = (K + 31) / 32;
    dim3 grid((NT + 3) / 4, (MT + 7) / 8);
    gemm_bf16_kernel<<<grid, 128, 0, stream>>>(Apk, Bpk, bias, C, MT, NT, KT, M, N, mode);
  };

  // ---- patch embedding ----
  conv1_kernel<<<(BATCH * 8 * 1800 + 255) / 256, 256, 0, stream>>>(src, cw1, cb1, c1);
  gn_gelu_kernel<<<BATCH, 256, 0, stream>>>(c1, gn1_g, gn1_b, 8, 1800, 14400);

  { // conv2 as GEMM: M=64*180, N=80, K=8*101=808
    int total = 720 * 26 * 512;
    im2col2_kernel<<<(total + 255) / 256, 256, 0, stream>>>(c1, Apk, total);
    packB(cw2, 80, 808);                  // cw2 [80,8,101] is [N,K] row-major
    gemm(cb2, c2, 11520, 80, 808, 0);
  }
  gn_gelu_kernel<<<BATCH, 256, 0, stream>>>(c2, gn2_g, gn2_b, 80, 1, 14400);

  { // conv3 as GEMM: M=64*18, N=800, K=80*51=4080 -> x in [b,s,d] layout
    int total = 72 * 128 * 512;
    im2col3_kernel<<<(total + 255) / 256, 256, 0, stream>>>(c2, Apk, total);
    packB(cw3, 800, 4080);                // cw3 [800,80,51] is [N,K] row-major
    gemm(cb3, x, 1152, 800, 4080, 0);
  }
  gn_gelu_kernel<<<BATCH, 256, 0, stream>>>(x, gn3_g, gn3_b, 800, 1, 14400);
  pe_add_kernel<<<(1152 * DIM + 255) / 256, 256, 0, stream>>>(x);

  // ---- transformer layers ----
  for (int i = 0; i < NLAY; ++i) {
    const float* wi  = ipw + (size_t)i * 3 * DIM * DIM;
    const float* bi  = ipb + (size_t)i * 3 * DIM;
    const float* wo  = aow + (size_t)i * DIM * DIM;
    const float* bo  = aob + (size_t)i * DIM;
    const float* w1  = l1w + (size_t)i * FFD * DIM;
    const float* b1  = l1b + (size_t)i * FFD;
    const float* w2  = l2w + (size_t)i * DIM * FFD;
    const float* b2  = l2b + (size_t)i * DIM;

    // QKV projection
    packA(x, 1152, DIM);
    packB(wi, 3 * DIM, DIM);
    gemm(bi, qkv, 1152, 3 * DIM, DIM, 0);

    // attention
    attn_kernel<<<BATCH * NHEAD, 128, 0, stream>>>(qkv, att);

    // attention out-proj
    packA(att, 1152, DIM);
    packB(wo, DIM, DIM);
    gemm(bo, tmp, 1152, DIM, DIM, 0);

    add_ln_kernel<<<1152, 256, 0, stream>>>(x, tmp, ln1g + (size_t)i * DIM, ln1b + (size_t)i * DIM);

    // FFN
    packA(x, 1152, DIM);
    packB(w1, FFD, DIM);
    gemm(b1, hbuf, 1152, FFD, DIM, 1);    // + gelu

    packA(hbuf, 1152, FFD);
    packB(w2, DIM, FFD);
    gemm(b2, tmp, 1152, DIM, FFD, 0);

    add_ln_kernel<<<1152, 256, 0, stream>>>(x, tmp, ln2g + (size_t)i * DIM, ln2b + (size_t)i * DIM);
  }

  // ---- final projection ----
  packA(x, 1152, DIM);
  packB(outw, DIM, DIM);
  gemm(outb, out, 1152, DIM, DIM, 0);

  (void)in_sizes; (void)n_in; (void)out_size; (void)ws_size;
}
